// SAGEConv_687194767735
// MI455X (gfx1250) — matrix operand order repaired and checked by reading, not compiled, not run on hardware
//
#include <hip/hip_runtime.h>
#include <hip/hip_bf16.h>

typedef __attribute__((ext_vector_type(16))) __bf16 v16bf;
typedef __attribute__((ext_vector_type(8)))  float  v8f;

#define D_IN   256
#define D_K    512
#define MAX_DEG 16
#define LDS_PITCH 520   // halves per row: 512 + 8 pad -> 1040B row stride, conflict-free b128 reads

__device__ __forceinline__ unsigned short f2bf(float f) {
    unsigned int u = __float_as_uint(f);
    unsigned int r = u + 0x7FFFu + ((u >> 16) & 1u);   // round-to-nearest-even
    return (unsigned short)(r >> 16);
}

union AFrag { v16bf v; uint4 q[2]; };

// W [256][512] f32  ->  Wb [512][256] bf16  (Wb[k][n] = W[n][k])
__global__ void sage_wprep(const float* __restrict__ W, unsigned short* __restrict__ Wb) {
    const int k = blockIdx.x;      // 0..511
    const int n = threadIdx.x;     // 0..255
    Wb[k * D_IN + n] = f2bf(W[n * D_K + k]);
}

__global__ __launch_bounds__(128)
void sage_fused(const float* __restrict__ x,
                const int* __restrict__ edge,
                const unsigned short* __restrict__ Wb,
                const float* __restrict__ bias,
                float* __restrict__ out,
                int n_nodes) {
    __shared__ unsigned short h_lds[16][LDS_PITCH];

    const int tid   = threadIdx.x;
    const int node0 = blockIdx.x * 16;

    // ---------- Phase A: build h = [x | mean(neigh)] (16 x 512) bf16 tile in LDS ----------
    {
        const int row   = tid >> 3;      // node within tile (0..15)
        const int chunk = tid & 7;       // 8 threads/node, 32 features each
        const int fbase = chunk * 32;
        const int node  = node0 + row;

        float4 sf[8], ag[8];
        #pragma unroll
        for (int i = 0; i < 8; ++i) {
            sf[i] = make_float4(0.f, 0.f, 0.f, 0.f);
            ag[i] = make_float4(0.f, 0.f, 0.f, 0.f);
        }

        if (node < n_nodes) {
            const float4* xs = (const float4*)(x + (size_t)node * D_IN + fbase);
            #pragma unroll
            for (int i = 0; i < 8; ++i) sf[i] = xs[i];

            int deg = 0;
            #pragma unroll
            for (int d = 0; d < MAX_DEG; ++d) {
                const int idx = edge[node * MAX_DEG + d];
                if (idx >= 0) {
                    ++deg;
                    const float4* xn = (const float4*)(x + (size_t)idx * D_IN + fbase);
                    #pragma unroll
                    for (int i = 0; i < 8; ++i) {
                        float4 v = xn[i];
                        ag[i].x += v.x; ag[i].y += v.y; ag[i].z += v.z; ag[i].w += v.w;
                    }
                }
            }
            const float inv = 1.0f / (float)(deg > 0 ? deg : 1);
            #pragma unroll
            for (int i = 0; i < 8; ++i) {
                ag[i].x *= inv; ag[i].y *= inv; ag[i].z *= inv; ag[i].w *= inv;
            }
        }

        unsigned int* dst_s = (unsigned int*)&h_lds[row][fbase];
        unsigned int* dst_a = (unsigned int*)&h_lds[row][D_IN + fbase];
        #pragma unroll
        for (int i = 0; i < 8; ++i) {
            dst_s[i * 2 + 0] = (unsigned int)f2bf(sf[i].x) | ((unsigned int)f2bf(sf[i].y) << 16);
            dst_s[i * 2 + 1] = (unsigned int)f2bf(sf[i].z) | ((unsigned int)f2bf(sf[i].w) << 16);
            dst_a[i * 2 + 0] = (unsigned int)f2bf(ag[i].x) | ((unsigned int)f2bf(ag[i].y) << 16);
            dst_a[i * 2 + 1] = (unsigned int)f2bf(ag[i].z) | ((unsigned int)f2bf(ag[i].w) << 16);
        }
    }
    __syncthreads();

    // ---------- Phase B: C(16x256) = A(16x512 bf16, LDS) x B(512x256 bf16, Wb) ----------
    const int lane = tid & 31;           // wave32
    const int wave = tid >> 5;           // 4 waves, 4 N-tiles each

    // A-frag addressing per ISA 16-bit A layout:
    //   lanes 0-15 : row = lane,    elems 0-7 = K+0..7,  elems 8-15 = K+16..23
    //   lanes 16-31: row = lane-16, elems 0-7 = K+8..15, elems 8-15 = K+24..31
    const int arow = lane & 15;
    const int koff = (lane < 16) ? 0 : 8;

    v8f acc[4];
    #pragma unroll
    for (int t = 0; t < 4; ++t) {
        const float bv = bias[(wave * 4 + t) * 16 + (lane & 15)];   // C: lane -> fixed column
        #pragma unroll
        for (int r = 0; r < 8; ++r) acc[t][r] = bv;
    }

    #pragma unroll 4
    for (int kk = 0; kk < 16; ++kk) {
        AFrag a;
        const uint4* asrc = (const uint4*)&h_lds[arow][kk * 32 + koff];
        a.q[0] = asrc[0];      // K + {0|8} .. +7
        a.q[1] = asrc[2];      // +16 halves -> K + {16|24} .. +7

        // B layout: lane -> column N = lane&15; elems 0-7 = K+{0|8}..+7, elems 8-15 = K+{16|24}..+7
        const int krow = kk * 32 + koff;
        const unsigned short* bbase = Wb + (size_t)krow * D_IN + (lane & 15);

        #pragma unroll
        for (int t = 0; t < 4; ++t) {
            const int ntile = wave * 4 + t;
            AFrag b;
            unsigned short* bs = (unsigned short*)&b;
            #pragma unroll
            for (int s = 0; s < 8; ++s) {
                bs[s]     = bbase[(size_t)s * D_IN + ntile * 16];
                bs[8 + s] = bbase[(size_t)(16 + s) * D_IN + ntile * 16];
            }
            acc[t] = __builtin_amdgcn_wmma_f32_16x16x32_bf16(
                false, a.v, false, b.v, (short)0, acc[t], false, false);
        }
    }

    // ---------- Phase C: store (C layout: lane 0-15 -> n, rows 0-7; lane 16-31 -> rows 8-15) ----------
    const int n_local = lane & 15;
    const int m_base  = (lane < 16) ? 0 : 8;
    #pragma unroll
    for (int t = 0; t < 4; ++t) {
        const int col = (wave * 4 + t) * 16 + n_local;
        #pragma unroll
        for (int r = 0; r < 8; ++r) {
            const int node = node0 + m_base + r;
            if (node < n_nodes)
                out[(size_t)node * D_IN + col] = acc[t][r];
        }
    }
}

extern "C" void kernel_launch(void* const* d_in, const int* in_sizes, int n_in,
                              void* d_out, int out_size, void* d_ws, size_t ws_size,
                              hipStream_t stream) {
    const float* x    = (const float*)d_in[0];
    const int*   edge = (const int*)d_in[1];
    const float* W    = (const float*)d_in[2];
    const float* bias = (const float*)d_in[3];
    float* out        = (float*)d_out;
    unsigned short* Wb = (unsigned short*)d_ws;   // 512*256*2 = 256 KB scratch

    const int n_nodes = in_sizes[0] / D_IN;

    sage_wprep<<<dim3(D_K), dim3(D_IN), 0, stream>>>(W, Wb);

    const int nblocks = (n_nodes + 15) / 16;
    sage_fused<<<dim3(nblocks), dim3(128), 0, stream>>>(x, edge, Wb, bias, out, n_nodes);
}
